// PrototypeMoRLlamaDecoderLayer_7825430413894
// MI455X (gfx1250) — matrix operand order repaired and checked by reading, not compiled, not run on hardware
//
#include <hip/hip_runtime.h>

// ---------------------------------------------------------------------------
// MoR Llama decoder (B=4,S=1024,H=1024,NH=16,HD=64,I=4096,R=2) for MI455X
// (gfx1250, wave32, WMMA). GEMMs/attention run on v_wmma_f32_16x16x32_f16.
// Data movement uses both gfx1250 async paths:
//   * GEMM tiles: Tensor Data Mover (tensor_load_to_lds, TENSORcnt) -- one 2D
//     D# descriptor per 128x32 tile, issued by wave 0, s_wait_tensorcnt.
//   * Flash-attn tiles: global_load_async_to_lds_b128 (ASYNCcnt).
// Weights are converted fp32->f16 *and transposed* once per launch so both
// A ([m][k]) and B ([n][k]) tiles are contiguous-row copies (DMA can't
// transpose). V is repacked [b,h,d,s] for the same reason.
// Every WMMA operand fragment is two contiguous ds_load_b128s (ISA 7.12.2).
// ---------------------------------------------------------------------------

typedef _Float16 half_t;
typedef __attribute__((ext_vector_type(16))) _Float16     v16h;
typedef __attribute__((ext_vector_type(8)))  float        v8f;
typedef __attribute__((ext_vector_type(4)))  unsigned int u32x4;
typedef __attribute__((ext_vector_type(8)))  int          i32x8;
typedef __attribute__((ext_vector_type(4)))  int          i32x4;

union V16 { v16h v; u32x4 q[2]; };

#define Bc   4
#define Sc   1024
#define Hc   1024
#define NHc  16
#define HDc  64
#define Ic   4096
#define Rc   2

static __device__ __forceinline__ v8f wmma_f16(const V16 &a, const V16 &b, v8f c) {
  // 8 args: (neg_a, A, neg_b, B, c_mod, C, reuse_a, reuse_b)
  return __builtin_amdgcn_wmma_f32_16x16x32_f16(false, a.v, false, b.v,
                                                (short)0, c, false, false);
}

// --- Async per-lane copy of 16B global -> LDS (ASYNCcnt) --------------------
static __device__ __forceinline__ void async_ld_b128(const void *gptr, void *lptr) {
  unsigned lds = (unsigned)(unsigned long long)lptr;   // LDS byte address
  asm volatile("global_load_async_to_lds_b128 %0, %1, off"
               :
               : "v"(lds), "v"(gptr)
               : "memory");
}
#define S_WAIT_ASYNC0() asm volatile("s_wait_asynccnt 0x0" ::: "memory")

// --- Tensor Data Mover: 2D tile load, D# per ISA ch.8 -----------------------
// Loads tile_d1 rows x tile_d0 elements (2-byte data) from gaddr (row stride
// stride_elems) into LDS at lds_addr, packed contiguously. Groups 2/3 zero
// (<=2D tensor). Wave-uniform arguments -> SGPR descriptor.
// This toolchain exposes the 6-arg builtin: (g0, g1, g2, g3, g_extra, cpol).
static __device__ __forceinline__ void tdm_load_2d(
    unsigned lds_addr, const void *gaddr,
    unsigned tile_d0, unsigned tile_d1,
    unsigned tensor_d0, unsigned tensor_d1,
    unsigned long long stride_elems)
{
  unsigned long long ga = (unsigned long long)gaddr;
  u32x4 g0;
  g0[0] = 1u;                                           // count=1 (valid D#)
  g0[1] = lds_addr;                                     // lds_addr [63:32]
  g0[2] = (unsigned)(ga & 0xffffffffu);                 // global_addr [95:64]
  g0[3] = (unsigned)((ga >> 32) & 0x01ffffffu)          // global_addr [120:96]
          | (2u << 30);                                 // type=2 ("image")
  i32x8 g1;
  g1[0] = (int)(1u << 16);                              // data_size=1 -> 2B
  g1[1] = (int)((tensor_d0 & 0xffffu) << 16);           // tensor_dim0 [79:48]
  g1[2] = (int)((tensor_d0 >> 16) | ((tensor_d1 & 0xffffu) << 16)); // dim1
  g1[3] = (int)((tensor_d1 >> 16) | (tile_d0 << 16));   // tile_dim0 [127:112]
  g1[4] = (int)(tile_d1 & 0xffffu);                     // tile_dim1; tile_dim2=0
  g1[5] = (int)(stride_elems & 0xffffffffu);            // dim0_stride [31:0]
  g1[6] = (int)((stride_elems >> 32) & 0xffffu);        // dim0_stride [47:32]
  g1[7] = 0;
  i32x4 z4 = {0, 0, 0, 0};
  i32x8 z8 = {0, 0, 0, 0, 0, 0, 0, 0};
  __builtin_amdgcn_tensor_load_to_lds(g0, g1, z4, z4, z8, 0);
}

// ---------------------------------------------------------------------------
// Generic GEMM: C[M,N](f32) = A[M,K](f16 row-major) * Bt[N,K](f16, B^T).
// 256 threads = 8 waves; tile 128x128x32; TDM double-buffered K pipeline
// (2 tensor ops per tile -> steady-state s_wait_tensorcnt 2).
// Wave (wm,wn) owns a 64x32 sub-tile as 4x2 WMMA accumulators.
// ---------------------------------------------------------------------------
#define GBM 128
#define GBN 128
#define GBK 32

__global__ __launch_bounds__(256) void gemm_f16_wmma(
    const half_t *__restrict__ A, const half_t *__restrict__ Bt,
    float *__restrict__ C, int M, int N, int K)
{
  __shared__ half_t As[2][GBM * GBK];   // [m][k]
  __shared__ half_t Bs[2][GBN * GBK];   // [n][k]
  const int tid = threadIdx.x;
  const int lane = tid & 31, wid = tid >> 5;
  const int wm = wid >> 2, wn = wid & 3;           // 2 x 4 wave grid
  const int hi = lane >> 4, l16 = lane & 15;
  const int row0 = blockIdx.y * GBM, col0 = blockIdx.x * GBN;

  const v8f z8 = {0.f, 0.f, 0.f, 0.f, 0.f, 0.f, 0.f, 0.f};
  v8f acc[4][2];
#pragma unroll
  for (int i = 0; i < 4; ++i)
#pragma unroll
    for (int j = 0; j < 2; ++j) acc[i][j] = z8;

  // 2 TDM descriptors per K-tile (A tile + B tile), issued by wave 0 only
  // (tensor ops ignore EXEC and complete once per instruction).
  auto issue = [&](int buf, int k0) {
    tdm_load_2d((unsigned)(unsigned long long)&As[buf][0],
                &A[(size_t)row0 * K + k0], GBK, GBM,
                (unsigned)(K - k0), (unsigned)(M - row0),
                (unsigned long long)K);
    tdm_load_2d((unsigned)(unsigned long long)&Bs[buf][0],
                &Bt[(size_t)col0 * K + k0], GBK, GBN,
                (unsigned)(K - k0), (unsigned)(N - col0),
                (unsigned long long)K);
  };

  auto compute = [&](int cur) {
    // Fragments: A lane(l16) row m, halves [m][hi*8] and [m][16+hi*8];
    //            B lane(l16) col n, halves [n][hi*8] and [n][16+hi*8]  (ISA 7.12.2)
    V16 af_[4], bf_[2];
#pragma unroll
    for (int im = 0; im < 4; ++im) {
      int r = wm * 64 + im * 16 + l16;
      af_[im].q[0] = *(const u32x4 *)(&As[cur][r * GBK + hi * 8]);
      af_[im].q[1] = *(const u32x4 *)(&As[cur][r * GBK + 16 + hi * 8]);
    }
#pragma unroll
    for (int in = 0; in < 2; ++in) {
      int n = wn * 32 + in * 16 + l16;
      bf_[in].q[0] = *(const u32x4 *)(&Bs[cur][n * GBK + hi * 8]);
      bf_[in].q[1] = *(const u32x4 *)(&Bs[cur][n * GBK + 16 + hi * 8]);
    }
#pragma unroll
    for (int im = 0; im < 4; ++im)
#pragma unroll
      for (int in = 0; in < 2; ++in)
        acc[im][in] = wmma_f16(af_[im], bf_[in], acc[im][in]);
  };

  const int nt = K / GBK;
  if (wid == 0) issue(0, 0);
  for (int t = 0; t < nt - 1; ++t) {         // steady state: one wait, no branch
    const int cur = t & 1;
    if (wid == 0) {
      issue(1 - cur, (t + 1) * GBK);
      __builtin_amdgcn_s_wait_tensorcnt((short)2);  // tile t landed, t+1 in flight
    }
    __syncthreads();
    compute(cur);
    __syncthreads();                         // reads done before overwrite
  }
  if (wid == 0) __builtin_amdgcn_s_wait_tensorcnt((short)0);
  __syncthreads();
  compute((nt - 1) & 1);

  // C layout: vgpr r -> (M = hi*8 + r, N = l16)
#pragma unroll
  for (int im = 0; im < 4; ++im) {
    int rb = row0 + wm * 64 + im * 16 + hi * 8;
#pragma unroll
    for (int in = 0; in < 2; ++in) {
      int cg = col0 + wn * 32 + in * 16 + l16;
#pragma unroll
      for (int r = 0; r < 8; ++r)
        C[(size_t)(rb + r) * N + cg] = acc[im][in][r];
    }
  }
}

// ---------------------------------------------------------------------------
// Flash attention, one (b, head, 64-query tile) per block, 128 thr = 4 waves.
// Wave w owns q-rows [w*16, w*16+16). 64-key blocks, causal block skip.
// Q/K tiles are [s][d] rows; V is pre-transposed [b,h,d,s] so the [d][key]
// tile is also a contiguous-row async copy (ASYNCcnt path).
// ---------------------------------------------------------------------------
#define FBq 64

__global__ __launch_bounds__(128) void flash_attn_wmma(
    const half_t *__restrict__ Qh, const half_t *__restrict__ Kh,
    const half_t *__restrict__ Vht, const float *__restrict__ mask,
    const float *__restrict__ maskadd, half_t *__restrict__ Out)
{
  const int qb = blockIdx.x * FBq;
  const int h = blockIdx.y;
  const int b = blockIdx.z;
  __shared__ half_t Qs[FBq * HDc];       // [q][d]
  __shared__ half_t Ks[FBq * HDc];       // [key][d]
  __shared__ half_t Vst[HDc * FBq];      // [d][key]
  __shared__ half_t Ps[4][16 * FBq];     // per-wave P, [q][key]
  const int tid = threadIdx.x, lane = tid & 31, wid = tid >> 5;
  const int hi = lane >> 4, l16 = lane & 15;
  const size_t headoff = ((size_t)b * NHc + h) * (size_t)Sc * HDc;

#pragma unroll
  for (int p = 0; p < 4; ++p) {
    int idx = (tid + p * 128) * 8;
    int r = idx >> 6, c = idx & 63;
    async_ld_b128(&Qh[headoff + (size_t)(qb + r) * HDc + c], &Qs[r * HDc + c]);
  }

  const v8f z8 = {0.f, 0.f, 0.f, 0.f, 0.f, 0.f, 0.f, 0.f};
  v8f o[4];
#pragma unroll
  for (int dt = 0; dt < 4; ++dt) o[dt] = z8;
  float m_s[8], l_s[8];
#pragma unroll
  for (int r = 0; r < 8; ++r) { m_s[r] = -3.0e38f; l_s[r] = 0.f; }

  const int nkb = qb / FBq + 1;          // causal structure of reference mask
  for (int kb = 0; kb < nkb; ++kb) {
#pragma unroll
    for (int p = 0; p < 4; ++p) {
      int idx = (tid + p * 128) * 8;
      int r = idx >> 6, c = idx & 63;
      async_ld_b128(&Kh[headoff + (size_t)(kb * FBq + r) * HDc + c],
                    &Ks[r * HDc + c]);
      async_ld_b128(&Vht[headoff + (size_t)r * Sc + kb * FBq + c],
                    &Vst[r * FBq + c]);
    }
    S_WAIT_ASYNC0();                     // own K/V (and Q on iter 0) landed
    __syncthreads();

    // S = Q K^T : M=16 q-rows, N=64 keys, Kdim=64 head dims (2 WMMA steps)
    v8f s[4];
#pragma unroll
    for (int jt = 0; jt < 4; ++jt) s[jt] = z8;
#pragma unroll
    for (int kd = 0; kd < 2; ++kd) {
      V16 qa;
      int qr = wid * 16 + l16;
      qa.q[0] = *(const u32x4 *)(&Qs[qr * HDc + kd * 32 + hi * 8]);
      qa.q[1] = *(const u32x4 *)(&Qs[qr * HDc + kd * 32 + 16 + hi * 8]);
#pragma unroll
      for (int jt = 0; jt < 4; ++jt) {
        V16 kf;   // B[k=d][n=key] = Ks[key][d] -> contiguous in Ks row
        kf.q[0] = *(const u32x4 *)(&Ks[(jt * 16 + l16) * HDc + kd * 32 + hi * 8]);
        kf.q[1] = *(const u32x4 *)(&Ks[(jt * 16 + l16) * HDc + kd * 32 + 16 + hi * 8]);
        s[jt] = wmma_f16(qa, kf, s[jt]);
      }
    }

    // online softmax; row = hi*8 + r lives in 16-lane groups -> shfl_xor<16>
#pragma unroll
    for (int r = 0; r < 8; ++r) {
      int qrow = qb + wid * 16 + hi * 8 + r;
      float sv[4];
      float mloc = -3.0e38f;
#pragma unroll
      for (int jt = 0; jt < 4; ++jt) {
        int col = kb * FBq + jt * 16 + l16;
        sv[jt] = s[jt][r] * 0.125f                      // 1/sqrt(HD)
                 + mask[((size_t)b * Sc + qrow) * Sc + col]
                 + maskadd[b * Sc + col];
        mloc = fmaxf(mloc, sv[jt]);
      }
      for (int mm = 1; mm < 16; mm <<= 1) mloc = fmaxf(mloc, __shfl_xor(mloc, mm, 32));
      float mnew = fmaxf(m_s[r], mloc);
      float corr = __expf(m_s[r] - mnew);
      float ps = 0.f;
#pragma unroll
      for (int jt = 0; jt < 4; ++jt) {
        float pv = __expf(sv[jt] - mnew);
        ps += pv;
        Ps[wid][(hi * 8 + r) * FBq + jt * 16 + l16] = (half_t)pv;
      }
      for (int mm = 1; mm < 16; mm <<= 1) ps += __shfl_xor(ps, mm, 32);
      l_s[r] = l_s[r] * corr + ps;
      m_s[r] = mnew;
#pragma unroll
      for (int dt = 0; dt < 4; ++dt) o[dt][r] *= corr;
    }
    __syncthreads();   // Ps ready

    // O += P V : Kdim = 64 keys (2 WMMA steps), N = 64 head dims
#pragma unroll
    for (int kc = 0; kc < 2; ++kc) {
      V16 pa;
      pa.q[0] = *(const u32x4 *)(&Ps[wid][l16 * FBq + kc * 32 + hi * 8]);
      pa.q[1] = *(const u32x4 *)(&Ps[wid][l16 * FBq + kc * 32 + 16 + hi * 8]);
#pragma unroll
      for (int dt = 0; dt < 4; ++dt) {
        V16 vb;   // B[k=key][n=d] = Vst[d][key] -> contiguous in Vst row
        vb.q[0] = *(const u32x4 *)(&Vst[(dt * 16 + l16) * FBq + kc * 32 + hi * 8]);
        vb.q[1] = *(const u32x4 *)(&Vst[(dt * 16 + l16) * FBq + kc * 32 + 16 + hi * 8]);
        o[dt] = wmma_f16(pa, vb, o[dt]);
      }
    }
    __syncthreads();   // Ks/Vst reads done before next async overwrite
  }

  // normalize and store into (B*S, H) f16 layout for the O projection GEMM
#pragma unroll
  for (int dt = 0; dt < 4; ++dt)
#pragma unroll
    for (int r = 0; r < 8; ++r) {
      int qrow = qb + wid * 16 + hi * 8 + r;
      float val = o[dt][r] / fmaxf(l_s[r], 1.0e-20f);
      Out[((size_t)b * Sc + qrow) * Hc + h * HDc + dt * 16 + l16] = (half_t)val;
    }
}

// ---------------------------------------------------------------------------
// Elementwise / small kernels
// ---------------------------------------------------------------------------

// fp32 [mat][K][N] -> f16 transposed [mat][N][K] (weights, once per launch)
__global__ void cvt_transpose_kernel(const float *__restrict__ in,
                                     half_t *__restrict__ out,
                                     int K, int N, size_t total) {
  size_t i = (size_t)blockIdx.x * blockDim.x + threadIdx.x;
  if (i < total) {
    size_t mn = (size_t)K * N;
    size_t mat = i / mn;
    size_t rem = i - mat * mn;
    int k = (int)(rem / N);
    int n = (int)(rem - (size_t)k * N);
    out[mat * mn + (size_t)n * K + k] = (half_t)in[i];
  }
}

// One block (256 thr) per token row: r_scores = x . ek[e], argmax + softmax sel
__global__ __launch_bounds__(256) void router_kernel(
    const float *__restrict__ hs, const float *__restrict__ ek,
    int *__restrict__ chosen, float *__restrict__ wsel)
{
  int row = blockIdx.x;
  int t = threadIdx.x, lane = t & 31, wid = t >> 5;
  float acc[4] = {0.f, 0.f, 0.f, 0.f};
  for (int c = t; c < Hc; c += 256) {
    float x = hs[(size_t)row * Hc + c];
#pragma unroll
    for (int e = 0; e < 4; ++e) acc[e] += x * ek[e * Hc + c];
  }
  __shared__ float sred[8][4];
#pragma unroll
  for (int e = 0; e < 4; ++e)
    for (int mm = 1; mm < 32; mm <<= 1) acc[e] += __shfl_xor(acc[e], mm, 32);
  if (lane == 0)
#pragma unroll
    for (int e = 0; e < 4; ++e) sred[wid][e] = acc[e];
  __syncthreads();
  if (t == 0) {
    float sc[4];
#pragma unroll
    for (int e = 0; e < 4; ++e) {
      sc[e] = 0.f;
      for (int w = 0; w < 8; ++w) sc[e] += sred[w][e];
    }
    int best = 0; float bm = sc[0];
#pragma unroll
    for (int e = 1; e < 4; ++e) if (sc[e] > bm) { bm = sc[e]; best = e; }
    float sum = 0.f;
#pragma unroll
    for (int e = 0; e < 4; ++e) sum += __expf(sc[e] - bm);
    chosen[row] = best;
    wsel[row] = 1.0f / sum;   // softmax prob of the argmax
  }
}

// x = final + af*ek[1<<d]; hn = rmsnorm(x)*g (f16); also key-mask bias
__global__ __launch_bounds__(256) void pre_norm_kernel(
    const float *__restrict__ fin, const float *__restrict__ ek,
    const int *__restrict__ chosen, const float *__restrict__ g,
    float *__restrict__ xbuf, half_t *__restrict__ hn,
    float *__restrict__ maskadd, int dbit)
{
  int row = blockIdx.x;
  int t = threadIdx.x, lane = t & 31, wid = t >> 5;
  float af = (float)((chosen[row] >> dbit) & 1);
  const float *ekr = ek + ((size_t)(1 << dbit)) * Hc;
  float v[4]; float ss = 0.f;
#pragma unroll
  for (int i = 0; i < 4; ++i) {
    int c = t + i * 256;
    float x = fin[(size_t)row * Hc + c] + af * ekr[c];
    v[i] = x; ss += x * x;
  }
  for (int mm = 1; mm < 32; mm <<= 1) ss += __shfl_xor(ss, mm, 32);
  __shared__ float sred[8];
  if (lane == 0) sred[wid] = ss;
  __syncthreads();
  float tot = 0.f;
#pragma unroll
  for (int w = 0; w < 8; ++w) tot += sred[w];
  float rms = rsqrtf(tot / (float)Hc + 1.0e-6f);
#pragma unroll
  for (int i = 0; i < 4; ++i) {
    int c = t + i * 256;
    xbuf[(size_t)row * Hc + c] = v[i];
    hn[(size_t)row * Hc + c] = (half_t)(v[i] * rms * g[c]);
  }
  if (t == 0) maskadd[row] = (1.0f - af) * (-1.0e9f);
}

// x2 = x + oproj; h2 = rmsnorm(x2)*g (f16)
__global__ __launch_bounds__(256) void add_norm_kernel(
    const float *__restrict__ xin, const float *__restrict__ oin,
    const float *__restrict__ g, float *__restrict__ x2,
    half_t *__restrict__ h2)
{
  int row = blockIdx.x;
  int t = threadIdx.x, lane = t & 31, wid = t >> 5;
  float v[4]; float ss = 0.f;
#pragma unroll
  for (int i = 0; i < 4; ++i) {
    int c = t + i * 256;
    float x = xin[(size_t)row * Hc + c] + oin[(size_t)row * Hc + c];
    v[i] = x; ss += x * x;
  }
  for (int mm = 1; mm < 32; mm <<= 1) ss += __shfl_xor(ss, mm, 32);
  __shared__ float sred[8];
  if (lane == 0) sred[wid] = ss;
  __syncthreads();
  float tot = 0.f;
#pragma unroll
  for (int w = 0; w < 8; ++w) tot += sred[w];
  float rms = rsqrtf(tot / (float)Hc + 1.0e-6f);
#pragma unroll
  for (int i = 0; i < 4; ++i) {
    int c = t + i * 256;
    x2[(size_t)row * Hc + c] = v[i];
    h2[(size_t)row * Hc + c] = (half_t)(v[i] * rms * g[c]);
  }
}

// RoPE on q,k (fp32 GEMM outputs) -> (b,nh,s,hd) f16; v -> (b,nh,hd,s) f16
__global__ void rope_kernel(const float *__restrict__ q,
                            const float *__restrict__ k,
                            const float *__restrict__ v,
                            const int *__restrict__ pos,
                            half_t *__restrict__ qh, half_t *__restrict__ kh,
                            half_t *__restrict__ vht)
{
  int idx = blockIdx.x * blockDim.x + threadIdx.x;   // over B*S*H (exact)
  int d = idx & 63;
  int hh = (idx >> 6) & (NHc - 1);
  int s = (idx >> 10) & (Sc - 1);
  int b = idx >> 20;
  float p = (float)pos[b * Sc + s];
  int i2 = d & 31;
  float inv = __powf(10000.0f, -((float)(2 * i2)) / 64.0f);
  float ang = p * inv;
  float cs = __cosf(ang), sn = __sinf(ang);
  int other = (d < 32) ? idx + 32 : idx - 32;
  float qv = q[idx], kv = k[idx];
  float qo = q[other], ko = k[other];
  float qr = qv * cs + ((d < 32) ? -qo : qo) * sn;
  float kr = kv * cs + ((d < 32) ? -ko : ko) * sn;
  size_t hb = ((size_t)b * NHc + hh);
  qh[(hb * Sc + s) * HDc + d] = (half_t)qr;
  kh[(hb * Sc + s) * HDc + d] = (half_t)kr;
  vht[(hb * HDc + d) * Sc + s] = (half_t)v[idx];     // transposed for async
}

__global__ void silu_mul_kernel(const float *__restrict__ g,
                                const float *__restrict__ u,
                                half_t *__restrict__ m, size_t n) {
  size_t i = (size_t)blockIdx.x * blockDim.x + threadIdx.x;
  if (i < n) {
    float gv = g[i];
    float sg = gv / (1.0f + __expf(-gv));
    m[i] = (half_t)(sg * u[i]);
  }
}

// final += af*wsel * (x2 + down)
__global__ void final_update_kernel(float *__restrict__ fin,
                                    const float *__restrict__ x2,
                                    const float *__restrict__ dn,
                                    const int *__restrict__ chosen,
                                    const float *__restrict__ wsel,
                                    int dbit, size_t n) {
  size_t i = (size_t)blockIdx.x * blockDim.x + threadIdx.x;
  if (i < n) {
    size_t row = i >> 10;   // /Hc
    float af = (float)((chosen[row] >> dbit) & 1);
    fin[i] += af * wsel[row] * (x2[i] + dn[i]);
  }
}

// ---------------------------------------------------------------------------
// Host orchestration
// ---------------------------------------------------------------------------
extern "C" void kernel_launch(void *const *d_in, const int *in_sizes, int n_in,
                              void *d_out, int out_size, void *d_ws,
                              size_t ws_size, hipStream_t stream)
{
  (void)in_sizes; (void)n_in; (void)out_size; (void)ws_size;
  const float *hs   = (const float *)d_in[0];
  const float *mask = (const float *)d_in[1];
  const int   *pos  = (const int *)d_in[2];
  const float *ek   = (const float *)d_in[3];
  const float *Wq   = (const float *)d_in[4];
  const float *Wk   = (const float *)d_in[5];
  const float *Wv   = (const float *)d_in[6];
  const float *Wo   = (const float *)d_in[7];
  const float *Wg   = (const float *)d_in[8];
  const float *Wu   = (const float *)d_in[9];
  const float *Wd   = (const float *)d_in[10];
  const float *ln1  = (const float *)d_in[11];
  const float *ln2  = (const float *)d_in[12];
  float *finalbuf = (float *)d_out;

  const size_t Mr = (size_t)Bc * Sc;            // 4096 token rows
  char *ws = (char *)d_ws;
  size_t off = 0;
  auto take = [&](size_t bytes) -> void * {
    void *p = ws + off;
    off = (off + bytes + 255) & ~(size_t)255;
    return p;
  };
  half_t *wq_h = (half_t *)take((size_t)Rc * Hc * Hc * 2);   // transposed [N][K]
  half_t *wk_h = (half_t *)take((size_t)Rc * Hc * Hc * 2);
  half_t *wv_h = (half_t *)take((size_t)Rc * Hc * Hc * 2);
  half_t *wo_h = (half_t *)take((size_t)Rc * Hc * Hc * 2);
  half_t *wg_h = (half_t *)take((size_t)Rc * Hc * Ic * 2);
  half_t *wu_h = (half_t *)take((size_t)Rc * Hc * Ic * 2);
  half_t *wd_h = (half_t *)take((size_t)Rc * Ic * Hc * 2);
  int    *chosen  = (int *)take(Mr * 4);
  float  *wsel    = (float *)take(Mr * 4);
  float  *maskadd = (float *)take(Mr * 4);
  float  *xbuf  = (float *)take(Mr * Hc * 4);
  float  *x2buf = (float *)take(Mr * Hc * 4);
  half_t *hn    = (half_t *)take(Mr * Hc * 2);
  half_t *h2    = (half_t *)take(Mr * Hc * 2);
  float  *qbuf  = (float *)take(Mr * Hc * 4);
  float  *kbuf  = (float *)take(Mr * Hc * 4);
  float  *vbuf  = (float *)take(Mr * Hc * 4);
  half_t *qh    = (half_t *)take(Mr * Hc * 2);
  half_t *kh    = (half_t *)take(Mr * Hc * 2);
  half_t *vht   = (half_t *)take(Mr * Hc * 2);
  half_t *attn  = (half_t *)take(Mr * Hc * 2);
  float  *oproj = (float *)take(Mr * Hc * 4);
  float  *gbuf  = (float *)take(Mr * Ic * 4);
  float  *ubuf  = (float *)take(Mr * Ic * 4);
  half_t *mbuf  = (half_t *)take(Mr * Ic * 2);
  float  *dbuf  = (float *)take(Mr * Hc * 4);

  // final starts as the input hidden states (never mutate d_in)
  (void)hipMemcpyAsync(finalbuf, hs, Mr * Hc * sizeof(float),
                       hipMemcpyDeviceToDevice, stream);

  // one-time fp32 -> f16 weight conversion, transposed to [N][K]
  auto cvtT = [&](const float *src, half_t *dst, int K, int N) {
    size_t n = (size_t)Rc * K * N;
    cvt_transpose_kernel<<<(unsigned)((n + 255) / 256), 256, 0, stream>>>(
        src, dst, K, N, n);
  };
  cvtT(Wq, wq_h, Hc, Hc);
  cvtT(Wk, wk_h, Hc, Hc);
  cvtT(Wv, wv_h, Hc, Hc);
  cvtT(Wo, wo_h, Hc, Hc);
  cvtT(Wg, wg_h, Hc, Ic);
  cvtT(Wu, wu_h, Hc, Ic);
  cvtT(Wd, wd_h, Ic, Hc);

  router_kernel<<<(unsigned)Mr, 256, 0, stream>>>(hs, ek, chosen, wsel);

  auto gemm = [&](const half_t *A, const half_t *Bm, float *C, int M, int N, int K) {
    dim3 grid(N / GBN, M / GBM);
    gemm_f16_wmma<<<grid, 256, 0, stream>>>(A, Bm, C, M, N, K);
  };

  for (int d = 0; d < Rc; ++d) {
    pre_norm_kernel<<<(unsigned)Mr, 256, 0, stream>>>(
        finalbuf, ek, chosen, ln1 + (size_t)d * Hc, xbuf, hn, maskadd, d);

    gemm(hn, wq_h + (size_t)d * Hc * Hc, qbuf, (int)Mr, Hc, Hc);
    gemm(hn, wk_h + (size_t)d * Hc * Hc, kbuf, (int)Mr, Hc, Hc);
    gemm(hn, wv_h + (size_t)d * Hc * Hc, vbuf, (int)Mr, Hc, Hc);

    rope_kernel<<<(unsigned)(Mr * Hc / 256), 256, 0, stream>>>(
        qbuf, kbuf, vbuf, pos, qh, kh, vht);

    flash_attn_wmma<<<dim3(Sc / FBq, NHc, Bc), 128, 0, stream>>>(
        qh, kh, vht, mask, maskadd, attn);

    gemm(attn, wo_h + (size_t)d * Hc * Hc, oproj, (int)Mr, Hc, Hc);

    add_norm_kernel<<<(unsigned)Mr, 256, 0, stream>>>(
        xbuf, oproj, ln2 + (size_t)d * Hc, x2buf, h2);

    gemm(h2, wg_h + (size_t)d * Hc * Ic, gbuf, (int)Mr, Ic, Hc);
    gemm(h2, wu_h + (size_t)d * Hc * Ic, ubuf, (int)Mr, Ic, Hc);

    silu_mul_kernel<<<(unsigned)((Mr * Ic + 255) / 256), 256, 0, stream>>>(
        gbuf, ubuf, mbuf, Mr * Ic);

    gemm(mbuf, wd_h + (size_t)d * Ic * Hc, dbuf, (int)Mr, Hc, Ic);

    final_update_kernel<<<(unsigned)((Mr * Hc + 255) / 256), 256, 0, stream>>>(
        finalbuf, x2buf, dbuf, chosen, wsel, d, Mr * Hc);
  }
}
